// AUGRULayer_6090263626303
// MI455X (gfx1250) — matrix operand order repaired and checked by reading, not compiled, not run on hardware
//
#include <hip/hip_runtime.h>

#define UDIM 128
#define TDIM 200
#define BDIM 2048
#define BM   64     // batch rows per workgroup
#define NTHREADS 256

typedef __attribute__((ext_vector_type(16))) __bf16 bf16x16;
typedef __attribute__((ext_vector_type(8)))  __bf16 bf16x8;
typedef __attribute__((ext_vector_type(8)))  float  f32x8;
typedef unsigned int u32;
typedef __attribute__((ext_vector_type(4)))  u32    u32x4;

union Frag  { bf16x16 bv; u32x4 q[2]; };
union Pack8 { bf16x8  b;  u32x4 q; };

static __device__ __forceinline__ float fast_sigmoid(float x) {
  return __builtin_amdgcn_rcpf(1.0f + __expf(-x));
}
static __device__ __forceinline__ float fast_tanh(float x) {
  return 1.0f - 2.0f * __builtin_amdgcn_rcpf(__expf(2.0f * x) + 1.0f);
}

#define WMMA_BF16(A, B, C) \
  __builtin_amdgcn_wmma_f32_16x16x32_bf16(false, (A), false, (B), (short)0, (C), false, false)

// A-fragment (16x32) of a transposed weight held row-major in LDS:
// lane row n (= base + lane&15), dwords0-7 <- k0..k0+15 (contiguous)
static __device__ __forceinline__ bf16x16 ldA(const __bf16* __restrict__ Wt, int n, int koff) {
  Frag f;
  const __bf16* p = Wt + n * UDIM + koff;       // koff = kf*32 + lh*16
  f.q[0] = *(const u32x4*)(p);
  f.q[1] = *(const u32x4*)(p + 8);
  return f.bv;
}

// B-fragment (32x16) of a transposed activation: lane = batch row m, 16 contiguous k.
static __device__ __forceinline__ bf16x16 ldB(const __bf16* __restrict__ act, int mloc, int koff) {
  Frag f;
  const __bf16* p = act + mloc * UDIM + koff;   // koff = kf*32 + lh*16
  f.q[0] = *(const u32x4*)(p);
  f.q[1] = *(const u32x4*)(p + 8);
  return f.bv;
}

__global__ __launch_bounds__(NTHREADS, 1)
void augru_fused_kernel(const float* __restrict__ X, const float* __restrict__ Att,
                        const float* __restrict__ Wxu, const float* __restrict__ bxu,
                        const float* __restrict__ Whu, const float* __restrict__ bhu,
                        const float* __restrict__ Wxr, const float* __restrict__ bxr,
                        const float* __restrict__ Whr, const float* __restrict__ bhr,
                        const float* __restrict__ Wxc, const float* __restrict__ bxc,
                        const float* __restrict__ Whc, const float* __restrict__ bhc,
                        float* __restrict__ Out) {
  extern __shared__ char smem[];
  __bf16* __restrict__ sW  = (__bf16*)smem;                                  // 6*128*128 bf16, W^T row-major
  float*  __restrict__ sB  = (float*)(smem + 6 * UDIM * UDIM * sizeof(__bf16)); // 3*128 f32 combined biases
  __bf16* __restrict__ sH  = (__bf16*)((char*)sB + 3 * UDIM * sizeof(float));   // BM*128 bf16 h, row-major [m][n]
  __bf16* __restrict__ sRH = sH + BM * UDIM;                                    // BM*128 bf16 r*h

  const int tid  = threadIdx.x;
  const int lane = tid & 31;
  const int wv   = tid >> 5;
  const int l15  = lane & 15;
  const int lh   = lane >> 4;
  const int nq   = wv & 3;    // n-quadrant: 32 hidden cols
  const int mh   = wv >> 2;   // m-half: 32 batch rows
  const int rowbase = blockIdx.x * BM;

  // ---------------- init: W^T -> LDS bf16, combined biases, h = 0
  {
    const float* Ws[6] = {Wxu, Whu, Wxr, Whr, Wxc, Whc};
#pragma unroll 1
    for (int m = 0; m < 6; ++m) {
      const float* w = Ws[m];
      __bf16* dst = sW + m * UDIM * UDIM;
      for (int i = tid; i < UDIM * UDIM; i += NTHREADS) {
        int k = i >> 7, n = i & (UDIM - 1);
        dst[n * UDIM + k] = (__bf16)w[i];
      }
    }
    for (int i = tid; i < UDIM; i += NTHREADS) {
      sB[0 * UDIM + i] = bxu[i] + bhu[i];
      sB[1 * UDIM + i] = bxr[i] + bhr[i];
      sB[2 * UDIM + i] = bxc[i] + bhc[i];
    }
    for (int i = tid; i < BM * UDIM; i += NTHREADS) sH[i] = (__bf16)0.0f;
  }

  const int n0 = nq * 32;                           // wave's hidden base
  const int nA = n0 + l15;                          // A-frag row for i=0 tile (i=1: +16)
  const int aOffA = lh * 16;                        // A-frag intra-kf offset
  const int aOffB = lh * 16;                        // B-frag intra-kf offset
  const int mA[2] = { mh * 32 + l15, mh * 32 + 16 + l15 };   // lane's local batch row per j
  const int nStore = n0 + lh * 8;                   // first n of this lane's 8 dwords (i=0)

  f32x8 accU[4], acc2[4], hReg[4];                  // index [i*2+j]
  bf16x16 xB[2][4];                                 // x B-frags, cached for 3 chains

#pragma unroll
  for (int q = 0; q < 4; ++q) hReg[q] = (f32x8){0.f, 0.f, 0.f, 0.f, 0.f, 0.f, 0.f, 0.f};

#pragma unroll 1
  for (int t = 0; t < TDIM; ++t) {
    __syncthreads();   // previous step's h stores visible

    // ---- x B-fragments: global fp32 -> bf16 (16 contiguous k per fragment)
#pragma unroll
    for (int j = 0; j < 2; ++j) {
      const float* xr = X + ((size_t)(rowbase + mA[j]) * TDIM + t) * UDIM;
#pragma unroll
      for (int kf = 0; kf < 4; ++kf) {
        const float* p = xr + kf * 32 + aOffB;
        bf16x16 b;
#pragma unroll
        for (int e = 0; e < 16; ++e) b[e] = (__bf16)p[e];
        xB[j][kf] = b;
      }
    }

    // ---- accumulators start from combined biases (vector LDS load, no broadcast)
#pragma unroll
    for (int i = 0; i < 2; ++i) {
      f32x8 bu = *(const f32x8*)(sB + 0 * UDIM + nStore + i * 16);
      f32x8 br = *(const f32x8*)(sB + 1 * UDIM + nStore + i * 16);
      accU[i * 2 + 0] = bu; accU[i * 2 + 1] = bu;
      acc2[i * 2 + 0] = br; acc2[i * 2 + 1] = br;
    }

    // ---- chain 1: accU += Wxu^T x^T ; acc2 += Wxr^T x^T   (A prefetched 1 kf ahead)
    {
      const __bf16* WU = sW + 0 * UDIM * UDIM;
      const __bf16* WR = sW + 2 * UDIM * UDIM;
      bf16x16 aU0 = ldA(WU, nA, aOffA),      aU1 = ldA(WU, nA + 16, aOffA);
      bf16x16 aR0 = ldA(WR, nA, aOffA),      aR1 = ldA(WR, nA + 16, aOffA);
#pragma unroll
      for (int kf = 0; kf < 4; ++kf) {
        bf16x16 nU0 = aU0, nU1 = aU1, nR0 = aR0, nR1 = aR1;
        if (kf < 3) {
          int ko = (kf + 1) * 32 + aOffA;
          nU0 = ldA(WU, nA, ko); nU1 = ldA(WU, nA + 16, ko);
          nR0 = ldA(WR, nA, ko); nR1 = ldA(WR, nA + 16, ko);
        }
        accU[0] = WMMA_BF16(aU0, xB[0][kf], accU[0]);
        accU[1] = WMMA_BF16(aU0, xB[1][kf], accU[1]);
        accU[2] = WMMA_BF16(aU1, xB[0][kf], accU[2]);
        accU[3] = WMMA_BF16(aU1, xB[1][kf], accU[3]);
        acc2[0] = WMMA_BF16(aR0, xB[0][kf], acc2[0]);
        acc2[1] = WMMA_BF16(aR0, xB[1][kf], acc2[1]);
        acc2[2] = WMMA_BF16(aR1, xB[0][kf], acc2[2]);
        acc2[3] = WMMA_BF16(aR1, xB[1][kf], acc2[3]);
        aU0 = nU0; aU1 = nU1; aR0 = nR0; aR1 = nR1;
      }
    }

    // ---- chain 2: accU += Whu^T h^T ; acc2 += Whr^T h^T   (h B-frags shared, prefetched)
    {
      const __bf16* WU = sW + 1 * UDIM * UDIM;
      const __bf16* WR = sW + 3 * UDIM * UDIM;
      bf16x16 hb0 = ldB(sH, mA[0], aOffB),   hb1 = ldB(sH, mA[1], aOffB);
      bf16x16 aU0 = ldA(WU, nA, aOffA),      aU1 = ldA(WU, nA + 16, aOffA);
      bf16x16 aR0 = ldA(WR, nA, aOffA),      aR1 = ldA(WR, nA + 16, aOffA);
#pragma unroll
      for (int kf = 0; kf < 4; ++kf) {
        bf16x16 nb0 = hb0, nb1 = hb1, nU0 = aU0, nU1 = aU1, nR0 = aR0, nR1 = aR1;
        if (kf < 3) {
          int koA = (kf + 1) * 32 + aOffA, koB = (kf + 1) * 32 + aOffB;
          nb0 = ldB(sH, mA[0], koB); nb1 = ldB(sH, mA[1], koB);
          nU0 = ldA(WU, nA, koA);    nU1 = ldA(WU, nA + 16, koA);
          nR0 = ldA(WR, nA, koA);    nR1 = ldA(WR, nA + 16, koA);
        }
        accU[0] = WMMA_BF16(aU0, hb0, accU[0]);
        accU[1] = WMMA_BF16(aU0, hb1, accU[1]);
        accU[2] = WMMA_BF16(aU1, hb0, accU[2]);
        accU[3] = WMMA_BF16(aU1, hb1, accU[3]);
        acc2[0] = WMMA_BF16(aR0, hb0, acc2[0]);
        acc2[1] = WMMA_BF16(aR0, hb1, acc2[1]);
        acc2[2] = WMMA_BF16(aR1, hb0, acc2[2]);
        acc2[3] = WMMA_BF16(aR1, hb1, acc2[3]);
        hb0 = nb0; hb1 = nb1; aU0 = nU0; aU1 = nU1; aR0 = nR0; aR1 = nR1;
      }
    }

    // ---- r = sigmoid(r_pre); rh = r * h -> one packed ds_store_b128 per tile
#pragma unroll
    for (int i = 0; i < 2; ++i) {
#pragma unroll
      for (int j = 0; j < 2; ++j) {
        Pack8 pk;
#pragma unroll
        for (int v = 0; v < 8; ++v) {
          float r = fast_sigmoid(acc2[i * 2 + j][v]);
          pk.b[v] = (__bf16)(r * hReg[i * 2 + j][v]);
        }
        *(u32x4*)(sRH + mA[j] * UDIM + nStore + i * 16) = pk.q;
      }
    }
    __syncthreads();   // rh visible across n-quadrant waves

    // attention (issue early; constant across dwords since a depends on m only)
    float av0 = Att[(size_t)(rowbase + mA[0]) * TDIM + t];
    float av1 = Att[(size_t)(rowbase + mA[1]) * TDIM + t];

    // ---- chain 3: acc2 = b_c + Wxc^T x^T + Whc^T (rh)^T
#pragma unroll
    for (int i = 0; i < 2; ++i) {
      f32x8 bc = *(const f32x8*)(sB + 2 * UDIM + nStore + i * 16);
      acc2[i * 2 + 0] = bc; acc2[i * 2 + 1] = bc;
    }
    {
      const __bf16* WX = sW + 4 * UDIM * UDIM;
      const __bf16* WH = sW + 5 * UDIM * UDIM;
      bf16x16 rb0 = ldB(sRH, mA[0], aOffB),  rb1 = ldB(sRH, mA[1], aOffB);
      bf16x16 aX0 = ldA(WX, nA, aOffA),      aX1 = ldA(WX, nA + 16, aOffA);
      bf16x16 aH0 = ldA(WH, nA, aOffA),      aH1 = ldA(WH, nA + 16, aOffA);
#pragma unroll
      for (int kf = 0; kf < 4; ++kf) {
        bf16x16 nb0 = rb0, nb1 = rb1, nX0 = aX0, nX1 = aX1, nH0 = aH0, nH1 = aH1;
        if (kf < 3) {
          int koA = (kf + 1) * 32 + aOffA, koB = (kf + 1) * 32 + aOffB;
          nb0 = ldB(sRH, mA[0], koB); nb1 = ldB(sRH, mA[1], koB);
          nX0 = ldA(WX, nA, koA);     nX1 = ldA(WX, nA + 16, koA);
          nH0 = ldA(WH, nA, koA);     nH1 = ldA(WH, nA + 16, koA);
        }
        acc2[0] = WMMA_BF16(aX0, xB[0][kf], acc2[0]);
        acc2[1] = WMMA_BF16(aX0, xB[1][kf], acc2[1]);
        acc2[2] = WMMA_BF16(aX1, xB[0][kf], acc2[2]);
        acc2[3] = WMMA_BF16(aX1, xB[1][kf], acc2[3]);
        acc2[0] = WMMA_BF16(aH0, rb0, acc2[0]);
        acc2[1] = WMMA_BF16(aH0, rb1, acc2[1]);
        acc2[2] = WMMA_BF16(aH1, rb0, acc2[2]);
        acc2[3] = WMMA_BF16(aH1, rb1, acc2[3]);
        rb0 = nb0; rb1 = nb1; aX0 = nX0; aX1 = nX1; aH0 = nH0; aH1 = nH1;
      }
    }

    // ---- combine: h' = (1 - a*u)*h + a*u*tanh(c); fp32 in hReg, bf16 packed to LDS
#pragma unroll
    for (int i = 0; i < 2; ++i) {
#pragma unroll
      for (int j = 0; j < 2; ++j) {
        float aj = j ? av1 : av0;
        Pack8 pk;
        f32x8 hn;
#pragma unroll
        for (int v = 0; v < 8; ++v) {
          float u  = fast_sigmoid(accU[i * 2 + j][v]);
          float c  = fast_tanh(acc2[i * 2 + j][v]);
          float au = aj * u;
          float h  = (1.0f - au) * hReg[i * 2 + j][v] + au * c;
          hn[v] = h;
          pk.b[v] = (__bf16)h;
        }
        hReg[i * 2 + j] = hn;
        *(u32x4*)(sH + mA[j] * UDIM + nStore + i * 16) = pk.q;
      }
    }
  }

  // ---- final h (fp32) -> Out [B, U]; lane writes 8 consecutive n per tile
#pragma unroll
  for (int i = 0; i < 2; ++i) {
#pragma unroll
    for (int j = 0; j < 2; ++j) {
      *(f32x8*)(Out + (size_t)(rowbase + mA[j]) * UDIM + nStore + i * 16) = hReg[i * 2 + j];
    }
  }
}

extern "C" void kernel_launch(void* const* d_in, const int* in_sizes, int n_in,
                              void* d_out, int out_size, void* d_ws, size_t ws_size,
                              hipStream_t stream) {
  (void)in_sizes; (void)n_in; (void)out_size; (void)d_ws; (void)ws_size;
  const float* X   = (const float*)d_in[0];
  const float* Att = (const float*)d_in[1];
  const float* Wxu = (const float*)d_in[2];
  const float* bxu = (const float*)d_in[3];
  const float* Whu = (const float*)d_in[4];
  const float* bhu = (const float*)d_in[5];
  const float* Wxr = (const float*)d_in[6];
  const float* bxr = (const float*)d_in[7];
  const float* Whr = (const float*)d_in[8];
  const float* bhr = (const float*)d_in[9];
  const float* Wxc = (const float*)d_in[10];
  const float* bxc = (const float*)d_in[11];
  const float* Whc = (const float*)d_in[12];
  const float* bhc = (const float*)d_in[13];
  float* Out = (float*)d_out;

  const size_t lds_bytes = (size_t)6 * UDIM * UDIM * sizeof(__bf16)
                         + (size_t)3 * UDIM * sizeof(float)
                         + (size_t)2 * BM * UDIM * sizeof(__bf16);
  (void)hipFuncSetAttribute((const void*)augru_fused_kernel,
                            hipFuncAttributeMaxDynamicSharedMemorySize,
                            (int)lds_bytes);

  dim3 grid(BDIM / BM);
  dim3 block(NTHREADS);
  augru_fused_kernel<<<grid, block, lds_bytes, stream>>>(
      X, Att, Wxu, bxu, Whu, bhu, Wxr, bxr, Whr, bhr, Wxc, bxc, Whc, bhc, Out);
}
